// GroupRopeAttention_1443109011690
// MI455X (gfx1250) — matrix operand order repaired and checked by reading, not compiled, not run on hardware
//
#include <hip/hip_runtime.h>
#include <math.h>

typedef __attribute__((ext_vector_type(16))) _Float16 v16h;
typedef __attribute__((ext_vector_type(8)))  _Float16 v8h;
typedef __attribute__((ext_vector_type(8)))  float    v8f;

// Problem constants (match reference)
constexpr int EMB = 1024;
constexpr int HD  = 128;
constexpr int NH  = 16;
constexpr int Bsz = 2;
constexpr int L   = 2048;
constexpr int BL  = Bsz * L;     // 4096 GEMM rows
constexpr int NQ  = NH * HD;     // 2048 q-proj cols

#define WMMA_F16(a, b, c) \
  __builtin_amdgcn_wmma_f32_16x16x32_f16(false, (a), false, (b), (short)0, (c), false, false)

// ---- WMMA fragment loaders (wave32, per CDNA5 ISA layouts) ----
// A-fragment 16x32 f16 from row-major [M][K]: lane holds row (lane&15),
// elements 0..7 at k = 8*half, elements 8..15 at k = 16 + 8*half.
static __device__ __forceinline__ v16h frag_a(const _Float16* row, int h) {
  v8h lo = *(const v8h*)(row + 8 * h);
  v8h hi = *(const v8h*)(row + 16 + 8 * h);
  v16h r;
#pragma unroll
  for (int i = 0; i < 8; ++i) { r[i] = lo[i]; r[i + 8] = hi[i]; }
  return r;
}
// B-fragment 32x16 f16 loaded from a TRANSPOSED source BT[N][K]:
// lane holds col n=(lane&15) -> row n of BT; elements 0..7 at k = 8*half,
// elements 8..15 at k = 16 + 8*half (same order as frag_a).
static __device__ __forceinline__ v16h frag_bt(const _Float16* row, int h) {
  v8h lo = *(const v8h*)(row + 8 * h);
  v8h hi = *(const v8h*)(row + 16 + 8 * h);
  v16h r;
#pragma unroll
  for (int i = 0; i < 8; ++i) { r[i] = lo[i]; r[i + 8] = hi[i]; }
  return r;
}

static __device__ __forceinline__ float red_max16(float v) {
#pragma unroll
  for (int off = 8; off; off >>= 1) v = fmaxf(v, __shfl_xor(v, off, 16));
  return v;
}
static __device__ __forceinline__ float red_sum16(float v) {
#pragma unroll
  for (int off = 8; off; off >>= 1) v += __shfl_xor(v, off, 16);
  return v;
}
static __device__ __forceinline__ void lds_turnaround_fence() {
  // In-wave LDS RAW: DS ops complete in order; fence compiler + wait dscnt.
  asm volatile("s_wait_dscnt 0" ::: "memory");
}

// gfx1250 async copy: global -> LDS, 16B per lane, tracked with ASYNCcnt.
// VDST = LDS byte address (low 32 bits of flat shared pointer), ADDR = 64-bit
// global address VGPR pair. (CDNA5 ISA op 98, CDNA5 ISA)
static __device__ __forceinline__ void async_load_b128(unsigned lds_off, const void* gptr) {
  asm volatile("global_load_async_to_lds_b128 %0, %1, off"
               :: "v"(lds_off), "v"(gptr) : "memory");
}
static __device__ __forceinline__ void wait_asynccnt0() {
  asm volatile("s_wait_asynccnt 0" ::: "memory");
}

// ---------------------------------------------------------------------------
// Kernel 1: fp32 -> f16 elementwise (x)
__global__ __launch_bounds__(256) void cvt_f16_kernel(const float* __restrict__ in,
                                                      _Float16* __restrict__ out, int n) {
  int i = blockIdx.x * 256 + threadIdx.x;
  if (i < n) out[i] = (_Float16)in[i];
}

// Kernel 2: transpose + convert: outT[n*K + k] = in[k*N + n]
__global__ __launch_bounds__(256) void cvt_transpose_kernel(const float* __restrict__ in,
                                                            _Float16* __restrict__ outT,
                                                            int K, int N) {
  int o = blockIdx.x * 256 + threadIdx.x;
  if (o >= K * N) return;
  int n = o / K;
  int k = o - n * K;
  outT[o] = (_Float16)in[(size_t)k * N + n];
}

// ---------------------------------------------------------------------------
// Kernel 3: fused QKV projection + RoPE. One wave per 16x16 C tile.
// grid.x = BL/16 (row tiles), grid.y = 128 (Q n-tiles) + 8 (K) + 8 (V) = 144
__global__ __launch_bounds__(32) void proj_rope_kernel(
    const _Float16* __restrict__ xh,
    const _Float16* __restrict__ WqT, const _Float16* __restrict__ WkT,
    const _Float16* __restrict__ WvT,
    _Float16* __restrict__ Qh,   // [b][h][l][d]
    _Float16* __restrict__ Kh,   // [b][l][d]
    _Float16* __restrict__ VT) { // [b][d][l]
  const int mt = blockIdx.x, nt = blockIdx.y;
  const int lane = threadIdx.x, lr = lane & 15, hs = lane >> 4;
  const int row0 = mt * 16;

  int mode, n0;
  const _Float16* bt;
  if (nt < 128)      { mode = 0; bt = WqT; n0 = nt * 16; }
  else if (nt < 136) { mode = 1; bt = WkT; n0 = (nt - 128) * 16; }
  else               { mode = 2; bt = WvT; n0 = (nt - 136) * 16; }

  const _Float16* arow = xh + (size_t)(row0 + lr) * EMB;
  const _Float16* brow = bt + (size_t)(n0 + lr) * EMB;

  v8f c = {};
#pragma unroll 4
  for (int kc = 0; kc < EMB; kc += 32) {
    v16h a = frag_a(arow + kc, hs);
    v16h b = frag_bt(brow + kc, hs);
    c = WMMA_F16(a, b, c);
  }

  // Epilogue: RoPE (Q, K) with interleaved pairs; V stored transposed.
  constexpr float LN_THETA_OVER_HALF = 9.210340371976184f / 64.0f; // ln(10000)/(HD/2)
#pragma unroll
  for (int r = 0; r < 8; ++r) {
    const int m = r + 8 * hs;
    const int row = row0 + m;
    const int b = row >> 11;         // / L
    const int ltok = row & (L - 1);  // % L
    float v = c[r];
    if (mode <= 1) {
      const int ncol = n0 + lr;
      const int d = (mode == 0) ? (ncol & (HD - 1)) : ncol;
      const float partner = __shfl_xor(v, 1, 32);   // adjacent column, same row
      const int kidx = d >> 1;
      const float freq = __expf(-(float)kidx * LN_THETA_OVER_HALF);
      const float ang = (float)ltok * freq;
      const float cs = cosf(ang), sn = sinf(ang);   // accurate range reduction
      const float o = (d & 1) ? (v * cs + partner * sn) : (v * cs - partner * sn);
      if (mode == 0) {
        const int hh = ncol >> 7;
        Qh[(((size_t)b * NH + hh) * L + ltok) * HD + d] = (_Float16)o;
      } else {
        Kh[((size_t)b * L + ltok) * HD + d] = (_Float16)o;
      }
    } else {
      const int d = n0 + lr;
      VT[((size_t)b * HD + d) * L + ltok] = (_Float16)v;
    }
  }
}

// ---------------------------------------------------------------------------
// Kernel 4: causal flash attention exploiting MQA K/V sharing.
// One 16-wave workgroup per (b, 16-row Q tile); wave w handles head w.
// K (32x128) and V^T (128x32) tiles are async-copied into double-buffered LDS
// and consumed by all 16 heads (16x K/V traffic reduction). Double buffering
// overlaps the ASYNCcnt copy of block j+1 with the WMMA work of block j and
// needs only ONE barrier per iteration.
// grid = (L/16, 1, Bsz), block = 512 threads (16 waves).
__global__ __launch_bounds__(512) void attn_kernel(
    const _Float16* __restrict__ Qh, const _Float16* __restrict__ Kh,
    const _Float16* __restrict__ VT, float* __restrict__ out) {
  __shared__ _Float16 ldsK[2][32 * HD];     // [buf][j_local][d]   2 x 8 KB
  __shared__ _Float16 ldsV[2][HD * 32];     // [buf][d][j_local]   2 x 8 KB
  __shared__ _Float16 ldsP[NH * 16 * 32];   // per-wave P tiles       16 KB

  const int tid = threadIdx.x;
  const int lane = tid & 31;
  const int h = tid >> 5;                   // wave id == head id
  const int lr = lane & 15, hs = lane >> 4;
  const int i0 = blockIdx.x * 16;
  const int b = blockIdx.z;

  const _Float16* qrow = Qh + (((size_t)b * NH + h) * L + i0 + lr) * HD;
  const _Float16* kbase = Kh + (size_t)b * L * HD;
  const _Float16* vbase = VT + (size_t)b * HD * L;

  v16h qa[4];
#pragma unroll
  for (int cix = 0; cix < 4; ++cix) qa[cix] = frag_a(qrow + 32 * cix, hs);

  const v8f vzero = {};
  v8f acc[8];
  float mrow[8], lrow[8];
  const float ninf = -__builtin_inff();
#pragma unroll
  for (int dc = 0; dc < 8; ++dc) acc[dc] = vzero;
#pragma unroll
  for (int r = 0; r < 8; ++r) { mrow[r] = ninf; lrow[r] = 0.0f; }

  // Cooperative async-copy assignments: 512 threads x 16B = one 8KB tile each.
  const int krow_ld = tid >> 4, kcol_ld = (tid & 15) * 8;   // K: 32 rows x 128
  const int vrow_ld = tid >> 2, vcol_ld = (tid & 3) * 8;    // V^T: 128 rows x 32
  unsigned kOff[2], vOff[2];
#pragma unroll
  for (int p = 0; p < 2; ++p) {
    kOff[p] = (unsigned)(uintptr_t)&ldsK[p][krow_ld * HD + kcol_ld];
    vOff[p] = (unsigned)(uintptr_t)&ldsV[p][vrow_ld * 32 + vcol_ld];
  }
  _Float16* pP = &ldsP[h * (16 * 32)];

  const float sm = 0.08838834764831845f; // 1/sqrt(128)
  const int jend = i0 + 16;              // causal bound

  // Prologue: start async copy of first K/V tile pair into buffer 0.
  async_load_b128(kOff[0], kbase + (size_t)krow_ld * HD + kcol_ld);
  async_load_b128(vOff[0], vbase + (size_t)vrow_ld * L + vcol_ld);

  int buf = 0;
  for (int j0 = 0; j0 < jend; j0 += 32, buf ^= 1) {
    // Only the current buffer's copies are outstanding: drain, then barrier.
    // The barrier also proves every wave finished reading buf^1 last iter,
    // so we can immediately start the next copy into buf^1.
    wait_asynccnt0();
    __syncthreads();
    if (j0 + 32 < jend) {
      async_load_b128(kOff[buf ^ 1],
                      kbase + (size_t)(j0 + 32 + krow_ld) * HD + kcol_ld);
      async_load_b128(vOff[buf ^ 1],
                      vbase + (size_t)vrow_ld * L + (j0 + 32) + vcol_ld);
    }
    const _Float16* Kt = &ldsK[buf][0];
    const _Float16* Vt = &ldsV[buf][0];

    // S = Q . K^T for two 16-column tiles, K from LDS
    v8f s0 = vzero, s1 = vzero;
#pragma unroll
    for (int cix = 0; cix < 4; ++cix) {
      v16h kb = frag_bt(&Kt[lr * HD + 32 * cix], hs);
      s0 = WMMA_F16(qa[cix], kb, s0);
    }
#pragma unroll
    for (int cix = 0; cix < 4; ++cix) {
      v16h kb = frag_bt(&Kt[(16 + lr) * HD + 32 * cix], hs);
      s1 = WMMA_F16(qa[cix], kb, s1);
    }

    // Online softmax per row (rows live across the 16 lanes of each half-wave)
#pragma unroll
    for (int r = 0; r < 8; ++r) {
      const int m = r + 8 * hs;
      const int irow = i0 + m;
      float a0 = (j0 + lr <= irow) ? s0[r] * sm : ninf;
      float a1 = (j0 + 16 + lr <= irow) ? s1[r] * sm : ninf;
      const float mx = red_max16(fmaxf(a0, a1));
      const float mnew = fmaxf(mrow[r], mx);
      const float corr = __expf(mrow[r] - mnew);
      mrow[r] = mnew;
      const float p0 = __expf(a0 - mnew);
      const float p1 = __expf(a1 - mnew);
      lrow[r] = lrow[r] * corr + red_sum16(p0 + p1);
#pragma unroll
      for (int dc = 0; dc < 8; ++dc) acc[dc][r] *= corr;
      pP[m * 32 + lr] = (_Float16)p0;
      pP[m * 32 + 16 + lr] = (_Float16)p1;
    }

    lds_turnaround_fence();

    // P (16x32) as A-fragment from LDS, then O += P . V (V from LDS)
    v16h p = frag_a(&pP[lr * 32], hs);
#pragma unroll
    for (int dc = 0; dc < 8; ++dc) {
      v16h vb = frag_bt(&Vt[(dc * 16 + lr) * 32], hs);
      acc[dc] = WMMA_F16(p, vb, acc[dc]);
    }
  }

  // Normalize and store: out[b][l][h*HD + d], fp32
  float* obase = out + ((size_t)b * L + i0) * NQ + h * HD;
#pragma unroll
  for (int r = 0; r < 8; ++r) {
    const int m = r + 8 * hs;
    const float inv = 1.0f / lrow[r];
#pragma unroll
    for (int dc = 0; dc < 8; ++dc)
      obase[(size_t)m * NQ + dc * 16 + lr] = acc[dc][r] * inv;
  }
}

// ---------------------------------------------------------------------------
extern "C" void kernel_launch(void* const* d_in, const int* in_sizes, int n_in,
                              void* d_out, int out_size, void* d_ws, size_t ws_size,
                              hipStream_t stream) {
  const float* x  = (const float*)d_in[0];
  const float* Wq = (const float*)d_in[1];
  const float* Wk = (const float*)d_in[2];
  const float* Wv = (const float*)d_in[3];
  float* out = (float*)d_out;

  // Workspace layout (f16 elements); total ~30.5 MB
  _Float16* ws  = (_Float16*)d_ws;
  _Float16* xh  = ws;                         // BL*EMB          = 4,194,304
  _Float16* WqT = xh  + (size_t)BL * EMB;     // NQ*EMB          = 2,097,152
  _Float16* WkT = WqT + (size_t)NQ * EMB;     // HD*EMB          =   131,072
  _Float16* WvT = WkT + (size_t)HD * EMB;     // HD*EMB          =   131,072
  _Float16* Qh  = WvT + (size_t)HD * EMB;     // B*NH*L*HD       = 8,388,608
  _Float16* Kh  = Qh  + (size_t)Bsz * NH * L * HD; // B*L*HD     =   524,288
  _Float16* VT  = Kh  + (size_t)Bsz * L * HD;      // B*HD*L     =   524,288

  // 1) convert x to f16
  {
    const int n = BL * EMB;
    cvt_f16_kernel<<<(n + 255) / 256, 256, 0, stream>>>(x, xh, n);
  }
  // 2) convert+transpose weights to f16 BT[N][K]
  {
    int n = EMB * NQ;
    cvt_transpose_kernel<<<(n + 255) / 256, 256, 0, stream>>>(Wq, WqT, EMB, NQ);
    n = EMB * HD;
    cvt_transpose_kernel<<<(n + 255) / 256, 256, 0, stream>>>(Wk, WkT, EMB, HD);
    cvt_transpose_kernel<<<(n + 255) / 256, 256, 0, stream>>>(Wv, WvT, EMB, HD);
  }
  // 3) fused QKV projection + RoPE (WMMA)
  {
    dim3 grid(BL / 16, NQ / 16 + HD / 16 + HD / 16); // 256 x 144
    proj_rope_kernel<<<grid, 32, 0, stream>>>(xh, WqT, WkT, WvT, Qh, Kh, VT);
  }
  // 4) causal flash attention: LDS-shared, double-buffered async K/V tiles
  {
    dim3 grid(L / 16, 1, Bsz);
    attn_kernel<<<grid, 512, 0, stream>>>(Qh, Kh, VT, out);
  }
}
